// Sender_6399501271454
// MI455X (gfx1250) — compile-verified, matrix operand order checked by reading, not run on hardware
//
#include <hip/hip_runtime.h>
#include <hip/hip_bf16.h>

// ---------------------------------------------------------------------------
// Gumbel-softmax LSTM decoder ("Sender") for MI455X / gfx1250.
//   BATCH=2048, HIDDEN=512, EMBED=256, VOCAB=1024, L=32 steps, TEMP=1.
//
// Roofline reasoning:
//  * GEMM work: ~307 GFLOP over 32 serial steps -> bf16 WMMA, fp32 accum.
//  * HBM floor: gumbel in (268MB) + probs out (277MB) ~ 24us @ 23.3TB/s.
//  * The real constraint is the per-step weight stream: each batch-tile
//    owner re-reads all weights (4.7MB bf16) from L2 every step. With
//    M=16 tiles each 1KB B-fragment feeds one WMMA -> L2-port bound.
//    Fix: BT=32 rows per workgroup, 2 row-tiles per wave, so every
//    B-fragment feeds 2 WMMAs -> halves L2 traffic (19GB -> 9.7GB) and
//    halves per-WGP port pressure. 64 workgroups x 8 wave32.
//  * LDS state (h/e/probs as bf16 A-operands) = 113KB -> dynamic LDS
//    (CDNA5 allows 320KB/workgroup); LSTM cell state c stays in fp32 regs.
//  * Softmax: logits+gumbel live in accumulator registers; row max/sum via
//    shfl_xor + tiny LDS cross-wave reduce. Next step's gumbel block is
//    prefetched (global_prefetch_b8) while GEMMs run.
// ---------------------------------------------------------------------------

#define VOCABSZ 1024
#define EMBEDSZ 256
#define HIDDENSZ 512
#define MAXLEN 32
#define BATCHSZ 2048
#define BT 32          // batch rows per workgroup (2 WMMA row-tiles)
#define RT 2           // row tiles per wave
#define NWAVE 8
#define WGSIZE 256

typedef __bf16 bf16_t;
typedef __attribute__((ext_vector_type(16))) __bf16 v16bf;
typedef __attribute__((ext_vector_type(8)))  float  v8f;

union BFrag { v16bf v; uint4 q[2]; };

// Load a 16x32 (A) or 32x16 (B) bf16 WMMA fragment. Caller passes the
// per-lane base pointer already offset by lane row/col and K-half:
//   elements 0..7  <- p[0..7]   (K = kbase .. kbase+7)
//   elements 8..15 <- p[16..23] (K = kbase+16 .. kbase+23)
__device__ __forceinline__ v16bf ldfrag(const bf16_t* p) {
  BFrag f;
  f.q[0] = *reinterpret_cast<const uint4*>(p);
  f.q[1] = *reinterpret_cast<const uint4*>(p + 16);
  return f.v;
}

__device__ __forceinline__ v8f wmma_bf16(v16bf a, v16bf b, v8f c) {
  return __builtin_amdgcn_wmma_f32_16x16x32_bf16(
      /*neg_a=*/false, a, /*neg_b=*/false, b,
      /*c_mod=*/(short)0, c, /*reuse_a=*/false, /*reuse_b=*/false);
}

__device__ __forceinline__ float sigmf(float x) {
  return 1.0f / (1.0f + __expf(-x));
}
__device__ __forceinline__ float tanh_fast(float x) {
  float a = fabsf(x);
  float e = __expf(-2.0f * a);
  float r = (1.0f - e) / (1.0f + e);
  return copysignf(r, x);
}

// ---------------------------------------------------------------------------
// Weight fp32 -> bf16 conversion into workspace (runs once per launch).
// ---------------------------------------------------------------------------
#define NIH  (4 * HIDDENSZ * EMBEDSZ)    // 524288
#define NHH  (4 * HIDDENSZ * HIDDENSZ)   // 1048576
#define NOUTW (VOCABSZ * HIDDENSZ)       // 524288
#define NEMBW (EMBEDSZ * VOCABSZ)        // 262144
#define NWTOT (NIH + NHH + NOUTW + NEMBW)

__global__ void convert_weights(const float* __restrict__ Wih,
                                const float* __restrict__ Whh,
                                const float* __restrict__ Wout,
                                const float* __restrict__ Wemb,
                                bf16_t* __restrict__ dst) {
  int idx = blockIdx.x * blockDim.x + threadIdx.x;
  if (idx >= NWTOT) return;
  float v;
  if (idx < NIH)                    v = Wih[idx];
  else if (idx < NIH + NHH)         v = Whh[idx - NIH];
  else if (idx < NIH + NHH + NOUTW) v = Wout[idx - NIH - NHH];
  else                              v = Wemb[idx - NIH - NHH - NOUTW];
  dst[idx] = (bf16_t)v;
}

// Dynamic LDS layout (bytes)
#define OFF_H   0
#define OFF_E   (OFF_H + BT * HIDDENSZ * 2)          // 32768
#define OFF_XS  (OFF_E + BT * EMBEDSZ * 2)           // +16384
#define OFF_RED (OFF_XS + BT * VOCABSZ * 2)          // +65536
#define OFF_RS  (OFF_RED + BT * NWAVE * 4)           // +1024
#define SMEM_BYTES (OFF_RS + BT * 4 + 16)            // ~115.9 KB

// ---------------------------------------------------------------------------
// Main recurrent kernel. grid = BATCH/BT = 64 workgroups of 256 threads.
// ---------------------------------------------------------------------------
__global__ __launch_bounds__(WGSIZE, 1)
void sender_kernel(const float* __restrict__ x,       // [B, H]
                   const float* __restrict__ gumbel,  // [L, B, V]
                   const float* __restrict__ b_ih,    // [4H]
                   const float* __restrict__ b_hh,    // [4H]
                   const float* __restrict__ b_out,   // [V]
                   const float* __restrict__ b_emb,   // [E]
                   const float* __restrict__ sos,     // [E]
                   const bf16_t* __restrict__ Wih,    // [4H, E] bf16
                   const bf16_t* __restrict__ Whh,    // [4H, H] bf16
                   const bf16_t* __restrict__ Wout,   // [V, H]  bf16
                   const bf16_t* __restrict__ Wemb,   // [E, V]  bf16
                   float* __restrict__ out)           // [B, L+1, V]
{
  extern __shared__ char smem[];
  bf16_t* Hlds    = (bf16_t*)(smem + OFF_H);    // [BT][HIDDEN]
  bf16_t* Elds    = (bf16_t*)(smem + OFF_E);    // [BT][EMBED]
  bf16_t* XSlds   = (bf16_t*)(smem + OFF_XS);   // [BT][VOCAB]
  float*  red     = (float*)(smem + OFF_RED);   // [BT][NWAVE]
  float*  rowstat = (float*)(smem + OFF_RS);    // [BT]

  const int tid  = threadIdx.x;
  const int lane = tid & 31;
  const int w    = tid >> 5;        // wave id 0..7
  const int b0   = blockIdx.x * BT;
  const int nn   = lane & 15;       // A row-in-tile / B column-in-tile
  const int hsel = lane >> 4;       // K-half (and C/D row-half) selector
  const int kh   = hsel * 8;        // K base offset within 32-wide block

  // ---- init: h0 = x (bf16), e0 = sos, eos frame of output ----
  for (int i = tid; i < BT * HIDDENSZ; i += WGSIZE) {
    int r = i / HIDDENSZ, c = i % HIDDENSZ;
    Hlds[i] = (bf16_t)x[(size_t)(b0 + r) * HIDDENSZ + c];
  }
  for (int i = tid; i < BT * EMBEDSZ; i += WGSIZE) {
    Elds[i] = (bf16_t)sos[i % EMBEDSZ];
  }
  for (int i = tid; i < BT * VOCABSZ; i += WGSIZE) {
    int r = i >> 10, c = i & (VOCABSZ - 1);
    out[((size_t)(b0 + r) * (MAXLEN + 1) + MAXLEN) * VOCABSZ + c] =
        (c == 0) ? 1.0f : 0.0f;
  }

  // LSTM cell state: 4 hidden col-tiles x 2 row-tiles, fp32, persistent.
  float c_reg[4][RT][8];
#pragma unroll
  for (int jt = 0; jt < 4; ++jt)
#pragma unroll
    for (int rt = 0; rt < RT; ++rt)
#pragma unroll
      for (int r = 0; r < 8; ++r) c_reg[jt][rt][r] = 0.0f;

  __syncthreads();

  for (int t = 0; t < MAXLEN; ++t) {
    // Prefetch next step's gumbel block while the GEMMs below run.
    if (t + 1 < MAXLEN) {
      const float* gnext = gumbel + ((size_t)(t + 1) * BATCHSZ + b0) * VOCABSZ;
#pragma unroll
      for (int i = 0; i < (BT * VOCABSZ) / (WGSIZE * 16); ++i)
        __builtin_prefetch(gnext + (size_t)(i * WGSIZE + tid) * 16, 0, 3);
    }

    // ================= Phase A: gates + LSTM cell ======================
    // wave w owns hidden column tiles w*4 .. w*4+3 (16 cols each).
    float hnew[4][RT][8];
#pragma unroll 1
    for (int jt = 0; jt < 4; ++jt) {
      const int n0  = (w * 4 + jt) * 16;
      const int col = n0 + nn;
      v8f acc[4][RT];  // i,f,g,o  x  row tiles
      {
        float bi  = b_ih[col]                + b_hh[col];
        float bff = b_ih[col +     HIDDENSZ] + b_hh[col +     HIDDENSZ];
        float bg  = b_ih[col + 2 * HIDDENSZ] + b_hh[col + 2 * HIDDENSZ];
        float bo  = b_ih[col + 3 * HIDDENSZ] + b_hh[col + 3 * HIDDENSZ];
#pragma unroll
        for (int rt = 0; rt < RT; ++rt)
#pragma unroll
          for (int r = 0; r < 8; ++r) {
            acc[0][rt][r] = bi;  acc[1][rt][r] = bff;
            acc[2][rt][r] = bg;  acc[3][rt][r] = bo;
          }
      }
      // e @ W_ih^T  (K = EMBED)
      {
        const bf16_t* pa0 = &Elds[nn * EMBEDSZ + kh];
        const bf16_t* pa1 = &Elds[(16 + nn) * EMBEDSZ + kh];
        const bf16_t* pw[4] = {
            Wih + (size_t)(col)*EMBEDSZ + kh,
            Wih + (size_t)(col +     HIDDENSZ) * EMBEDSZ + kh,
            Wih + (size_t)(col + 2 * HIDDENSZ) * EMBEDSZ + kh,
            Wih + (size_t)(col + 3 * HIDDENSZ) * EMBEDSZ + kh};
#pragma unroll 2
        for (int kb = 0; kb < EMBEDSZ; kb += 32) {
          v16bf a0 = ldfrag(pa0 + kb);
          v16bf a1 = ldfrag(pa1 + kb);
#pragma unroll
          for (int g = 0; g < 4; ++g) {
            v16bf b = ldfrag(pw[g] + kb);      // 1KB frag feeds 2 WMMAs
            acc[g][0] = wmma_bf16(a0, b, acc[g][0]);
            acc[g][1] = wmma_bf16(a1, b, acc[g][1]);
          }
        }
      }
      // h @ W_hh^T  (K = HIDDEN)
      {
        const bf16_t* pa0 = &Hlds[nn * HIDDENSZ + kh];
        const bf16_t* pa1 = &Hlds[(16 + nn) * HIDDENSZ + kh];
        const bf16_t* pw[4] = {
            Whh + (size_t)(col)*HIDDENSZ + kh,
            Whh + (size_t)(col +     HIDDENSZ) * HIDDENSZ + kh,
            Whh + (size_t)(col + 2 * HIDDENSZ) * HIDDENSZ + kh,
            Whh + (size_t)(col + 3 * HIDDENSZ) * HIDDENSZ + kh};
#pragma unroll 2
        for (int kb = 0; kb < HIDDENSZ; kb += 32) {
          v16bf a0 = ldfrag(pa0 + kb);
          v16bf a1 = ldfrag(pa1 + kb);
#pragma unroll
          for (int g = 0; g < 4; ++g) {
            v16bf b = ldfrag(pw[g] + kb);
            acc[g][0] = wmma_bf16(a0, b, acc[g][0]);
            acc[g][1] = wmma_bf16(a1, b, acc[g][1]);
          }
        }
      }
      // LSTM elementwise (C/D layout: element r of row-tile rt -> row
      // r + 8*hsel + 16*rt).
#pragma unroll
      for (int rt = 0; rt < RT; ++rt)
#pragma unroll
        for (int r = 0; r < 8; ++r) {
          float ig = sigmf(acc[0][rt][r]);
          float fg = sigmf(acc[1][rt][r]);
          float gg = tanh_fast(acc[2][rt][r]);
          float og = sigmf(acc[3][rt][r]);
          float cc = fg * c_reg[jt][rt][r] + ig * gg;
          c_reg[jt][rt][r] = cc;
          hnew[jt][rt][r]  = og * tanh_fast(cc);
        }
    }
    __syncthreads();   // everyone finished reading Hlds/Elds
#pragma unroll
    for (int jt = 0; jt < 4; ++jt)
#pragma unroll
      for (int rt = 0; rt < RT; ++rt)
#pragma unroll
        for (int r = 0; r < 8; ++r)
          Hlds[(r + 8 * hsel + 16 * rt) * HIDDENSZ + (w * 4 + jt) * 16 + nn] =
              (bf16_t)hnew[jt][rt][r];
    __syncthreads();   // h_new visible

    // ================= Phase B: logits + gumbel + softmax ==============
    // wave w owns vocab tiles w*8 .. w*8+7 ; logits stay in registers.
    v8f lac[8][RT];
    const int v0 = w * 128;
#pragma unroll
    for (int vt = 0; vt < 8; ++vt) {
      float bb = b_out[v0 + vt * 16 + nn];
#pragma unroll
      for (int rt = 0; rt < RT; ++rt)
#pragma unroll
        for (int r = 0; r < 8; ++r) lac[vt][rt][r] = bb;
    }
    {
      const bf16_t* pa0 = &Hlds[nn * HIDDENSZ + kh];
      const bf16_t* pa1 = &Hlds[(16 + nn) * HIDDENSZ + kh];
#pragma unroll 1
      for (int kb = 0; kb < HIDDENSZ; kb += 32) {
        v16bf a0 = ldfrag(pa0 + kb);
        v16bf a1 = ldfrag(pa1 + kb);
#pragma unroll
        for (int vt = 0; vt < 8; ++vt) {
          const bf16_t* pb =
              Wout + (size_t)(v0 + vt * 16 + nn) * HIDDENSZ + kh + kb;
          v16bf b = ldfrag(pb);
          lac[vt][0] = wmma_bf16(a0, b, lac[vt][0]);
          lac[vt][1] = wmma_bf16(a1, b, lac[vt][1]);
        }
      }
    }
    // add gumbel noise (TEMP = 1)
    {
      const float* gp = gumbel + ((size_t)t * BATCHSZ + b0) * VOCABSZ;
#pragma unroll
      for (int vt = 0; vt < 8; ++vt)
#pragma unroll
        for (int rt = 0; rt < RT; ++rt)
#pragma unroll
          for (int r = 0; r < 8; ++r)
            lac[vt][rt][r] += gp[(size_t)(r + 8 * hsel + 16 * rt) * VOCABSZ +
                                 v0 + vt * 16 + nn];
    }
    // row max: per-lane over 8 tiles, shfl over 16 lanes, LDS across waves
    float pmax[RT][8];
#pragma unroll
    for (int rt = 0; rt < RT; ++rt)
#pragma unroll
      for (int r = 0; r < 8; ++r) {
        float m = lac[0][rt][r];
#pragma unroll
        for (int vt = 1; vt < 8; ++vt) m = fmaxf(m, lac[vt][rt][r]);
        pmax[rt][r] = m;
      }
    for (int off = 1; off < 16; off <<= 1)
#pragma unroll
      for (int rt = 0; rt < RT; ++rt)
#pragma unroll
        for (int r = 0; r < 8; ++r)
          pmax[rt][r] = fmaxf(pmax[rt][r], __shfl_xor(pmax[rt][r], off, 32));
    if (nn == 0)
#pragma unroll
      for (int rt = 0; rt < RT; ++rt)
#pragma unroll
        for (int r = 0; r < 8; ++r)
          red[(r + 8 * hsel + 16 * rt) * NWAVE + w] = pmax[rt][r];
    __syncthreads();
    if (tid < BT) {
      float m = red[tid * NWAVE];
#pragma unroll
      for (int j = 1; j < NWAVE; ++j) m = fmaxf(m, red[tid * NWAVE + j]);
      rowstat[tid] = m;
    }
    __syncthreads();
    float rmax[RT][8];
#pragma unroll
    for (int rt = 0; rt < RT; ++rt)
#pragma unroll
      for (int r = 0; r < 8; ++r)
        rmax[rt][r] = rowstat[r + 8 * hsel + 16 * rt];
    // exp + row sum
    float psum[RT][8];
#pragma unroll
    for (int rt = 0; rt < RT; ++rt)
#pragma unroll
      for (int r = 0; r < 8; ++r) psum[rt][r] = 0.0f;
#pragma unroll
    for (int vt = 0; vt < 8; ++vt)
#pragma unroll
      for (int rt = 0; rt < RT; ++rt)
#pragma unroll
        for (int r = 0; r < 8; ++r) {
          float e = __expf(lac[vt][rt][r] - rmax[rt][r]);
          lac[vt][rt][r] = e;
          psum[rt][r] += e;
        }
    for (int off = 1; off < 16; off <<= 1)
#pragma unroll
      for (int rt = 0; rt < RT; ++rt)
#pragma unroll
        for (int r = 0; r < 8; ++r)
          psum[rt][r] += __shfl_xor(psum[rt][r], off, 32);
    if (nn == 0)
#pragma unroll
      for (int rt = 0; rt < RT; ++rt)
#pragma unroll
        for (int r = 0; r < 8; ++r)
          red[(r + 8 * hsel + 16 * rt) * NWAVE + w] = psum[rt][r];
    __syncthreads();
    if (tid < BT) {
      float s = 0.0f;
#pragma unroll
      for (int j = 0; j < NWAVE; ++j) s += red[tid * NWAVE + j];
      rowstat[tid] = s;
    }
    __syncthreads();
    float rinv[RT][8];
#pragma unroll
    for (int rt = 0; rt < RT; ++rt)
#pragma unroll
      for (int r = 0; r < 8; ++r)
        rinv[rt][r] = 1.0f / rowstat[r + 8 * hsel + 16 * rt];
    // write probabilities: fp32 to global out, bf16 to LDS for next GEMM
#pragma unroll
    for (int vt = 0; vt < 8; ++vt)
#pragma unroll
      for (int rt = 0; rt < RT; ++rt)
#pragma unroll
        for (int r = 0; r < 8; ++r) {
          float xs = lac[vt][rt][r] * rinv[rt][r];
          int M = r + 8 * hsel + 16 * rt;
          int cv = v0 + vt * 16 + nn;
          out[((size_t)(b0 + M) * (MAXLEN + 1) + t) * VOCABSZ + cv] = xs;
          XSlds[M * VOCABSZ + cv] = (bf16_t)xs;
        }
    __syncthreads();   // XS ready

    // ================= Phase C: e_new = xs @ W_emb^T + b_emb ===========
    // wave w owns embed tiles w*2, w*2+1 ; K = VOCAB.
    v8f eac[2][RT];
    const int e0 = w * 32;
#pragma unroll
    for (int et = 0; et < 2; ++et) {
      float bb = b_emb[e0 + et * 16 + nn];
#pragma unroll
      for (int rt = 0; rt < RT; ++rt)
#pragma unroll
        for (int r = 0; r < 8; ++r) eac[et][rt][r] = bb;
    }
    {
      const bf16_t* pa0 = &XSlds[nn * VOCABSZ + kh];
      const bf16_t* pa1 = &XSlds[(16 + nn) * VOCABSZ + kh];
#pragma unroll 2
      for (int kb = 0; kb < VOCABSZ; kb += 32) {
        v16bf a0 = ldfrag(pa0 + kb);
        v16bf a1 = ldfrag(pa1 + kb);
#pragma unroll
        for (int et = 0; et < 2; ++et) {
          const bf16_t* pb =
              Wemb + (size_t)(e0 + et * 16 + nn) * VOCABSZ + kh + kb;
          v16bf b = ldfrag(pb);
          eac[et][0] = wmma_bf16(a0, b, eac[et][0]);
          eac[et][1] = wmma_bf16(a1, b, eac[et][1]);
        }
      }
    }
#pragma unroll
    for (int et = 0; et < 2; ++et)
#pragma unroll
      for (int rt = 0; rt < RT; ++rt)
#pragma unroll
        for (int r = 0; r < 8; ++r)
          Elds[(r + 8 * hsel + 16 * rt) * EMBEDSZ + e0 + et * 16 + nn] =
              (bf16_t)eac[et][rt][r];
    __syncthreads();   // E ready for next step
  }
}

// ---------------------------------------------------------------------------
extern "C" void kernel_launch(void* const* d_in, const int* in_sizes, int n_in,
                              void* d_out, int out_size, void* d_ws,
                              size_t ws_size, hipStream_t stream) {
  const float* x      = (const float*)d_in[0];
  const float* gumbel = (const float*)d_in[1];
  const float* W_ih   = (const float*)d_in[2];
  const float* W_hh   = (const float*)d_in[3];
  const float* b_ih   = (const float*)d_in[4];
  const float* b_hh   = (const float*)d_in[5];
  const float* W_out  = (const float*)d_in[6];
  const float* b_out  = (const float*)d_in[7];
  const float* W_emb  = (const float*)d_in[8];
  const float* b_emb  = (const float*)d_in[9];
  const float* sos    = (const float*)d_in[10];

  bf16_t* wbf = (bf16_t*)d_ws;   // needs NWTOT*2 = 4.72 MB of workspace

  // Allow >64KB dynamic LDS (CDNA5: up to 320KB per workgroup).
  hipFuncSetAttribute(reinterpret_cast<const void*>(sender_kernel),
                      hipFuncAttributeMaxDynamicSharedMemorySize, SMEM_BYTES);

  convert_weights<<<(NWTOT + 255) / 256, 256, 0, stream>>>(
      W_ih, W_hh, W_out, W_emb, wbf);

  sender_kernel<<<BATCHSZ / BT, WGSIZE, SMEM_BYTES, stream>>>(
      x, gumbel, b_ih, b_hh, b_out, b_emb, sos,
      wbf,                       // W_ih bf16
      wbf + NIH,                 // W_hh bf16
      wbf + NIH + NHH,           // W_out bf16
      wbf + NIH + NHH + NOUTW,   // W_emb bf16
      (float*)d_out);
}
